// InformationExtractor_29068338659878
// MI455X (gfx1250) — matrix operand order repaired and checked by reading, not compiled, not run on hardware
//
#include <hip/hip_runtime.h>

// ---------- types ----------
typedef __attribute__((ext_vector_type(16))) __bf16 bf16x16;
typedef __attribute__((ext_vector_type(8)))  float  v8f;
typedef unsigned int  u32;
typedef unsigned short u16;
typedef __attribute__((ext_vector_type(4))) u32 uint4v;

union ABFrag {
    uint4v u[2];   // 32 bytes = 16 bf16
    bf16x16 v;
};

static __device__ __forceinline__ u16 f2bf(float f) {
    u32 u = __float_as_uint(f);
    u32 r = (u + 0x7FFFu + ((u >> 16) & 1u)) >> 16;   // round-to-nearest-even
    return (u16)r;
}

// CDNA5 async global->LDS copy, 16 bytes per lane (ASYNCcnt-tracked).
static __device__ __forceinline__ void async_copy_b128(u32 lds_off, const void* gptr) {
    asm volatile("global_load_async_to_lds_b128 %0, %1, off"
                 :: "v"(lds_off), "v"(gptr) : "memory");
}
static __device__ __forceinline__ void wait_asynccnt0() {
    asm volatile("s_wait_asynccnt 0" ::: "memory");
}

// =====================================================================
// Kernel W: repack conv2 weights [64][32][3][3] f32 (OIHW) into the
// per-lane bf16 B-fragment layout (contiguous K, same as A) this file uses for V_WMMA_F32_16X16X32_BF16:
//   bfrag[t][nw][lane][j]:  lane L holds N = nw*16 + (L%16),
//   K = (L/16)*16 + j  (input channel within tap t)
// =====================================================================
__global__ void repack_w2_kernel(const float* __restrict__ w2,
                                 u16* __restrict__ bfrag) {
    int i = blockIdx.x * 256 + threadIdx.x;
    if (i >= 9 * 4 * 32 * 16) return;
    int j    = i & 15;
    int lane = (i >> 4) & 31;
    int nw   = (i >> 9) & 3;
    int t    = i >> 11;
    int oc   = nw * 16 + (lane & 15);
    int ic   = (lane >> 4) * 16 + j;
    float v  = w2[((oc * 32 + ic) * 3 + t / 3) * 3 + (t % 3)];
    bfrag[i] = f2bf(v);
}

// =====================================================================
// Kernel A: conv1(1->32, 3x3 SAME) + bias + relu + 2x2 maxpool.
// One thread per pooled pixel; 32 bf16 channels stored channel-innermost.
// =====================================================================
__global__ void conv1_pool_kernel(const float* __restrict__ x,
                                  const float* __restrict__ w1,
                                  const float* __restrict__ b1,
                                  u16* __restrict__ h, int total) {
    __shared__ float lw[288];
    __shared__ float lb[32];
    int tid = threadIdx.x;
    if (tid < 288) lw[tid] = w1[tid];
    if (tid < 32)  lb[tid] = b1[tid];
    __syncthreads();

    int idx = blockIdx.x * 256 + tid;
    if (idx >= total) return;
    int b  = idx / 196;
    int r  = idx - b * 196;
    int oy = r / 14;
    int ox = r - oy * 14;

    float win[4][4];
    int iy0 = 2 * oy - 1;
    int ix0 = 2 * ox - 1;
    const float* xb = x + (size_t)b * 784;
    #pragma unroll
    for (int dy = 0; dy < 4; ++dy) {
        int iy = iy0 + dy;
        #pragma unroll
        for (int dx = 0; dx < 4; ++dx) {
            int ix = ix0 + dx;
            bool ok = (iy >= 0) & (iy < 28) & (ix >= 0) & (ix < 28);
            win[dy][dx] = ok ? xb[iy * 28 + ix] : 0.0f;
        }
    }

    union { u16 s[32]; uint4v u[4]; } hv;
    #pragma unroll
    for (int ic = 0; ic < 32; ++ic) {
        float s00 = 0.f, s01 = 0.f, s10 = 0.f, s11 = 0.f;
        #pragma unroll
        for (int ky = 0; ky < 3; ++ky) {
            #pragma unroll
            for (int kx = 0; kx < 3; ++kx) {
                float wv = lw[ic * 9 + ky * 3 + kx];
                s00 = fmaf(win[ky][kx],         wv, s00);
                s01 = fmaf(win[ky][kx + 1],     wv, s01);
                s10 = fmaf(win[ky + 1][kx],     wv, s10);
                s11 = fmaf(win[ky + 1][kx + 1], wv, s11);
            }
        }
        float m = fmaxf(fmaxf(s00, s01), fmaxf(s10, s11));
        hv.s[ic] = f2bf(fmaxf(m + lb[ic], 0.0f));
    }
    uint4v* dst = (uint4v*)(h + (size_t)idx * 32);
    dst[0] = hv.u[0]; dst[1] = hv.u[1]; dst[2] = hv.u[2]; dst[3] = hv.u[3];
}

// =====================================================================
// Kernel B: conv2(32->64) implicit GEMM, bf16 WMMA, fused bias/relu/pool.
//   block = 128 threads = 4 waves; each block: one pooled row (7 quads)
//   of 4 consecutive images -> 7 M-tiles of 16 rows; wave w: N=16w..16w+15.
//   B-fragments (9 taps) hoisted into registers across all 7 quads.
//   A-tiles staged via GLOBAL_LOAD_ASYNC_TO_LDS_B128 with ping-pong LDS
//   buffers; A-fragment ds reads software-pipelined (2-deep register
//   ping-pong) so each WMMA overlaps the next step's LDS loads.
// =====================================================================
__global__ void conv2_wmma_kernel(const u16* __restrict__ h,
                                  const u16* __restrict__ bfrag,
                                  const float* __restrict__ b2,
                                  float* __restrict__ feats) {
    __shared__ __align__(16) u16 ldsA[2][16 * 288];   // 2 x 9216 B ping-pong

    int tid  = threadIdx.x;
    int qrow = blockIdx.x % 7;            // pooled row py
    int grp  = blockIdx.x / 7;
    int b0   = grp * 4;

    // ---- stage A-tile for quad (qrow, px) into buffer buf ----
    auto stageA = [&](int buf, int px) {
        u16* dstbuf = &ldsA[buf][0];
        for (int seg = tid; seg < 144; seg += 128) {
            int m = seg / 9;               // row: image (m/4), quad pixel (m%4)
            int t = seg - m * 9;           // tap
            int b = b0 + (m >> 2);
            int p = m & 3;
            int cy = 2 * qrow + (p >> 1);
            int cx = 2 * px + (p & 1);
            int yy = cy + t / 3 - 1;
            int xx = cx + t % 3 - 1;
            u16* dst = dstbuf + m * 288 + t * 32;
            if (yy >= 0 && yy < 14 && xx >= 0 && xx < 14) {
                const u16* src = h + ((size_t)(b * 14 + yy) * 14 + xx) * 32;
                u32 loff = (u32)(size_t)dst;
                #pragma unroll
                for (int jj = 0; jj < 4; ++jj)
                    async_copy_b128(loff + jj * 16, src + jj * 8);
            } else {
                uint4v z = {0u, 0u, 0u, 0u};
                uint4v* d4 = (uint4v*)dst;
                d4[0] = z; d4[1] = z; d4[2] = z; d4[3] = z;
            }
        }
    };

    int wid  = tid >> 5;                   // wave id -> n_base = 16*wid
    int lane = tid & 31;
    int mrow = lane & 15;                  // A: M = lane%16
    int hi   = lane >> 4;
    int kb   = hi * 16;                    // A: K base within 32-wide step
    int n    = wid * 16 + (lane & 15);
    float bias = b2[n];

    // ---- hoist the 9 B-fragments into registers (reused for 7 quads) ----
    ABFrag barr[9];
    #pragma unroll
    for (int t = 0; t < 9; ++t) {
        const uint4v* pb =
            (const uint4v*)(bfrag + ((size_t)(t * 4 + wid) * 32 + lane) * 16);
        barr[t].u[0] = pb[0];
        barr[t].u[1] = pb[1];
    }

    auto loadA = [&](ABFrag& a, const u16* abuf, int t) {
        const u16* arow = abuf + mrow * 288 + t * 32 + kb;
        a.u[0] = *(const uint4v*)(arow);        // K = kb..kb+7
        a.u[1] = *(const uint4v*)(arow + 8);    // K = kb+8..kb+15
    };

    stageA(0, 0);
    wait_asynccnt0();
    __syncthreads();

    for (int qi = 0; qi < 7; ++qi) {
        int cur = qi & 1;
        if (qi < 6) stageA(cur ^ 1, qi + 1);   // async; overlaps WMMAs below

        const u16* abuf = &ldsA[cur][0];
        ABFrag aa[2];
        loadA(aa[0], abuf, 0);                 // prologue of A-read pipeline
        v8f c = {0.f, 0.f, 0.f, 0.f, 0.f, 0.f, 0.f, 0.f};
        #pragma unroll
        for (int t = 0; t < 9; ++t) {
            if (t < 8) loadA(aa[(t + 1) & 1], abuf, t + 1);  // prefetch next
            c = __builtin_amdgcn_wmma_f32_16x16x32_bf16(
                    false, aa[t & 1].v, false, barr[t].v, (short)0, c,
                    false, false);
        }

        // epilogue: bias + relu + within-lane 2x2 maxpool, store feats[B][49][64]
        int q = qrow * 7 + qi;
        float m0 = fmaxf(fmaxf(c[0], c[1]), fmaxf(c[2], c[3]));
        float m1 = fmaxf(fmaxf(c[4], c[5]), fmaxf(c[6], c[7]));
        float p0 = fmaxf(m0 + bias, 0.0f);
        float p1 = fmaxf(m1 + bias, 0.0f);
        int bA = b0 + 2 * hi;
        feats[((size_t)bA * 49 + q) * 64 + n]       = p0;
        feats[((size_t)(bA + 1) * 49 + q) * 64 + n] = p1;

        wait_asynccnt0();      // own wave's async stores of next buffer done
        __syncthreads();       // everyone done reading cur / staging next
    }
}

// =====================================================================
// Kernel C: attention (sigmoid gate) + 64 tiny experts + final head.
// =====================================================================
__global__ void tail_kernel(const float* __restrict__ feats,
                            const float* __restrict__ attn_w,
                            const float* __restrict__ attn_b,
                            const float* __restrict__ expert_w,
                            const float* __restrict__ expert_b,
                            const float* __restrict__ final_w,
                            const float* __restrict__ final_b,
                            float* __restrict__ out) {
    __shared__ float lf[49 * 64];
    __shared__ float lexp[128];
    int b = blockIdx.x;
    int e = threadIdx.x;   // channel 0..63

    for (int qv = 0; qv < 49; ++qv)
        lf[qv * 64 + e] = feats[((size_t)b * 49 + qv) * 64 + e];
    __syncthreads();

    float al = attn_b[0];
    float e0 = 0.f, e1 = 0.f;
    const float* ew = expert_w + (size_t)e * 49 * 2;
    #pragma unroll 7
    for (int qv = 0; qv < 49; ++qv) {
        float f = lf[qv * 64 + e];
        al = fmaf(f, attn_w[qv], al);
        e0 = fmaf(f, ew[qv * 2 + 0], e0);
        e1 = fmaf(f, ew[qv * 2 + 1], e1);
    }
    float a = 1.0f / (1.0f + expf(-al));
    lexp[e * 2 + 0] = fmaxf((e0 + expert_b[e * 2 + 0]) * a, 0.0f);
    lexp[e * 2 + 1] = fmaxf((e1 + expert_b[e * 2 + 1]) * a, 0.0f);
    __syncthreads();

    if (e < 2) {
        float acc = final_b[e];
        #pragma unroll 8
        for (int j = 0; j < 128; ++j)
            acc = fmaf(lexp[j], final_w[j * 2 + e], acc);
        out[(size_t)b * 2 + e] = acc;
    }
}

// =====================================================================
extern "C" void kernel_launch(void* const* d_in, const int* in_sizes, int n_in,
                              void* d_out, int out_size, void* d_ws, size_t ws_size,
                              hipStream_t stream) {
    const float* x        = (const float*)d_in[0];
    const float* conv1_w  = (const float*)d_in[1];
    const float* conv1_b  = (const float*)d_in[2];
    const float* conv2_w  = (const float*)d_in[3];
    const float* conv2_b  = (const float*)d_in[4];
    const float* attn_w   = (const float*)d_in[5];
    const float* attn_b   = (const float*)d_in[6];
    const float* expert_w = (const float*)d_in[7];
    const float* expert_b = (const float*)d_in[8];
    const float* final_w  = (const float*)d_in[9];
    const float* final_b  = (const float*)d_in[10];
    float* out = (float*)d_out;

    int B = in_sizes[0] / 784;                  // 8192

    // workspace layout (256B aligned sections)
    char* ws = (char*)d_ws;
    u16*  bfrag = (u16*)ws;                                  // 36,864 B
    u16*  h     = (u16*)(ws + 36864);                        // B*12544 B
    float* feats = (float*)(ws + 36864 + (size_t)B * 12544); // B*12544 B

    repack_w2_kernel<<<(9 * 4 * 32 * 16 + 255) / 256, 256, 0, stream>>>(conv2_w, bfrag);

    int total = B * 196;
    conv1_pool_kernel<<<(total + 255) / 256, 256, 0, stream>>>(x, conv1_w, conv1_b, h, total);

    conv2_wmma_kernel<<<7 * (B / 4), 128, 0, stream>>>(h, bfrag, conv2_b, feats);

    tail_kernel<<<B, 64, 0, stream>>>(feats, attn_w, attn_b, expert_w, expert_b,
                                      final_w, final_b, out);
}
